// MPNN_80333068304733
// MI455X (gfx1250) — matrix operand order repaired and checked by reading, not compiled, not run on hardware
//
#include <hip/hip_runtime.h>
#include <hip/hip_bf16.h>
#include <math.h>

// ---------------- problem constants ----------------
#define NN 30000      // nodes
#define EE 100000     // edges
#define GG 2048       // graphs
#define HH 64         // hidden
#define RR 1024       // readout feats
#define KFUSED 1088   // 16*64 (x (x) h outer product) + 64 (h for bias term)
#define NCHUNK (KFUSED / 32)
#define LSTRIDE 40    // padded LDS row stride (elems): 80B, 16B-aligned, bank-spread

typedef __attribute__((ext_vector_type(16))) __bf16 v16bf;
typedef __attribute__((ext_vector_type(8)))  __bf16 v8bf;
typedef __attribute__((ext_vector_type(8)))  float  v8f;

// ---------------- small device helpers ----------------
__device__ __forceinline__ float sigm(float x) { return 1.0f / (1.0f + __expf(-x)); }

// order-preserving float->uint encoding for atomicMax on floats
__device__ __forceinline__ unsigned fenc(float f) {
    unsigned u = __float_as_uint(f);
    return (u & 0x80000000u) ? ~u : (u | 0x80000000u);
}
__device__ __forceinline__ float fdec(unsigned u) {
    return __uint_as_float((u & 0x80000000u) ? (u & 0x7fffffffu) : ~u);
}

// A-fragment k permutation: fragment wants k-order {0..7,16..23 | 8..15,24..31};
// swaps aligned k-groups of 8, so aligned groups of 4 remain contiguous.
__device__ __forceinline__ int kperm(int k) {
    return (k & 7) | ((k & 8) << 1) | ((k & 16) >> 1);
}

// 32B fragment load as two 16B LDS vector loads (ds_load_b128), 16B-aligned by construction
__device__ __forceinline__ v16bf ld_frag(const __bf16* p) {
    union { v16bf v; v8bf h[2]; } u;
    u.h[0] = *(const v8bf*)(p);
    u.h[1] = *(const v8bf*)(p + 8);
    return u.v;
}

// pack 4 floats -> 4 bf16 and store as one 8B LDS store (dest 8B-aligned by construction)
__device__ __forceinline__ void st4bf(__bf16* d, float x, float y, float z, float w) {
    union { __bf16 b[4]; unsigned long long u; } pk;
    pk.b[0] = (__bf16)x; pk.b[1] = (__bf16)y; pk.b[2] = (__bf16)z; pk.b[3] = (__bf16)w;
    *(unsigned long long*)d = pk.u;
}

#define WMMA_BF16(a, b, c) \
    __builtin_amdgcn_wmma_f32_16x16x32_bf16(false, (a), false, (b), (short)0, (c), false, false)

// ---------------- utility kernels ----------------
__global__ void k_zero(float* __restrict__ p, long n) {
    long i = (long)blockIdx.x * blockDim.x + threadIdx.x;
    if (i < n) p[i] = 0.0f;
}

__global__ void k_copy(float* __restrict__ dst, const float* __restrict__ src, long n) {
    long i = (long)blockIdx.x * blockDim.x + threadIdx.x;
    if (i < n) dst[i] = src[i];
}

// Build Wbig in chunk-transposed layout: wt[kc][o][kk] = Wbig_row(kh=kc*32+kk, col o)
// where Wbig rows kh<1024 come from edge_W[k=kh/64][(h=kh%64)*64+o], else edge_b[(kh-1024)*64+o].
// This makes the edge kernel's B staging a straight 32B/thread copy.
__global__ void k_build_wbig(const float* __restrict__ eW, const float* __restrict__ eb,
                             __bf16* __restrict__ wt) {
    int i = blockIdx.x * blockDim.x + threadIdx.x;
    if (i >= KFUSED * 64) return;
    int kc = i >> 11;            // / 2048
    int rem = i & 2047;
    int o = rem >> 5, kk = rem & 31;
    int kh = kc * 32 + kk;
    float v;
    if (kh < 1024) { int k = kh >> 6, h = kh & 63; v = eW[k * 4096 + h * 64 + o]; }
    else           { int h = kh - 1024;            v = eb[h * 64 + o]; }
    wt[i] = (__bf16)v;
}

// ---------------- generic bf16 WMMA GEMM ----------------
// C[M,N] = act( A[M,K] @ B + bias (+ C if accumulate) )
// transB==0: B is [K,N] row-major. transB==1: B is [N,K] row-major (A @ B^T).
// act: 0 none, 1 relu, 2 prelu(alpha_p[0])
// Block = 128 threads (4 waves), tile 128(M) x 64(N); wave tile 32x64 => 8 v_wmma per K step.
__global__ void __launch_bounds__(128, 1)
k_gemm(const float* __restrict__ A, const float* __restrict__ B,
       const float* __restrict__ bias, float* __restrict__ C,
       int M, int N, int K, int transB, int act,
       const float* __restrict__ alpha_p, int accumulate) {
    __shared__ __bf16 Ap[128 * LSTRIDE];  // A tile, natural k order
    __shared__ __bf16 Bt[64 * LSTRIDE];   // transposed B tile (col-major by k)

    const int tid = threadIdx.x;
    const int m0 = blockIdx.y * 128, n0 = blockIdx.x * 64;
    const int wv = tid >> 5, lane = tid & 31, hi = lane >> 4, mr = lane & 15;

    v8f acc[2][4];
#pragma unroll
    for (int f = 0; f < 2; ++f)
#pragma unroll
        for (int t = 0; t < 4; ++t)
#pragma unroll
            for (int r = 0; r < 8; ++r) acc[f][t][r] = 0.0f;

    for (int k0 = 0; k0 < K; k0 += 32) {
        // ---- stage A tile (128x32) in natural k order, float4 loads + packed 8B stores
        for (int i = tid; i < 128 * 8; i += 128) {
            int r = i >> 3, c4 = (i & 7) * 4;
            int gm = m0 + r, gk = k0 + c4;
            float x = 0.f, y = 0.f, z = 0.f, w = 0.f;
            if (gm < M) {
                if (gk + 3 < K) {
                    float4 v = *(const float4*)&A[(size_t)gm * K + gk];
                    x = v.x; y = v.y; z = v.z; w = v.w;
                } else {
                    const float* ar = &A[(size_t)gm * K];
                    if (gk     < K) x = ar[gk];
                    if (gk + 1 < K) y = ar[gk + 1];
                    if (gk + 2 < K) z = ar[gk + 2];
                    if (gk + 3 < K) w = ar[gk + 3];
                }
            }
            st4bf(Ap + r * LSTRIDE + c4, x, y, z, w);
        }
        // ---- stage B tile (32x64) transposed into Bt[n][k]
        if (transB) {
            for (int i = tid; i < 64 * 8; i += 128) {
                int n = i >> 3, k4 = (i & 7) * 4;
                int gn = n0 + n, gk = k0 + k4;
                float x = 0.f, y = 0.f, z = 0.f, w = 0.f;
                if (gn < N) {
                    if (gk + 3 < K) {
                        float4 v = *(const float4*)&B[(size_t)gn * K + gk];
                        x = v.x; y = v.y; z = v.z; w = v.w;
                    } else {
                        const float* br = &B[(size_t)gn * K];
                        if (gk     < K) x = br[gk];
                        if (gk + 1 < K) y = br[gk + 1];
                        if (gk + 2 < K) z = br[gk + 2];
                        if (gk + 3 < K) w = br[gk + 3];
                    }
                }
                st4bf(Bt + n * LSTRIDE + k4, x, y, z, w);
            }
        } else {
            for (int i = tid; i < 32 * 64; i += 128) {
                int r = i >> 6, c = i & 63;   // r=k, c=n
                int gk = k0 + r, gn = n0 + c;
                float v = (gk < K && gn < N) ? B[(size_t)gk * N + gn] : 0.0f;
                Bt[c * LSTRIDE + r] = (__bf16)v;
            }
        }
        // prefetch next K chunk (global_prefetch_b8; speculative OOB prefetch is dropped)
        if (k0 + 32 < K) {
            int gm = m0 + tid;
            if (gm < M) __builtin_prefetch(&A[(size_t)gm * K + k0 + 32], 0, 1);
        }
        __syncthreads();

        const __bf16* arow = Ap + (size_t)(wv * 32) * LSTRIDE + hi * 16;
        v16bf a0 = ld_frag(arow + (size_t)mr * LSTRIDE);
        v16bf a1 = ld_frag(arow + (size_t)(16 + mr) * LSTRIDE);
        v16bf b0 = ld_frag(Bt + (size_t)( 0 + mr) * LSTRIDE + hi * 16);
        v16bf b1 = ld_frag(Bt + (size_t)(16 + mr) * LSTRIDE + hi * 16);
        v16bf b2 = ld_frag(Bt + (size_t)(32 + mr) * LSTRIDE + hi * 16);
        v16bf b3 = ld_frag(Bt + (size_t)(48 + mr) * LSTRIDE + hi * 16);

        acc[0][0] = WMMA_BF16(a0, b0, acc[0][0]);
        acc[0][1] = WMMA_BF16(a0, b1, acc[0][1]);
        acc[0][2] = WMMA_BF16(a0, b2, acc[0][2]);
        acc[0][3] = WMMA_BF16(a0, b3, acc[0][3]);
        acc[1][0] = WMMA_BF16(a1, b0, acc[1][0]);
        acc[1][1] = WMMA_BF16(a1, b1, acc[1][1]);
        acc[1][2] = WMMA_BF16(a1, b2, acc[1][2]);
        acc[1][3] = WMMA_BF16(a1, b3, acc[1][3]);
        __syncthreads();
    }

    float alpha = (act == 2 && alpha_p) ? alpha_p[0] : 0.0f;
#pragma unroll
    for (int f = 0; f < 2; ++f) {
#pragma unroll
        for (int t = 0; t < 4; ++t) {
#pragma unroll
            for (int r = 0; r < 8; ++r) {
                int gm = m0 + wv * 32 + f * 16 + r + 8 * hi;
                int gn = n0 + t * 16 + mr;
                if (gm < M && gn < N) {
                    float v = acc[f][t][r];
                    if (bias) v += bias[gn];
                    if (accumulate) v += C[(size_t)gm * N + gn];
                    if (act == 1)      v = v > 0.0f ? v : 0.0f;
                    else if (act == 2) v = v >= 0.0f ? v : alpha * v;
                    C[(size_t)gm * N + gn] = v;
                }
            }
        }
    }
}

// ---------------- fused NNConv message kernel ----------------
// m[e,:] = (x[e] (x) h[src[e]] , h[src[e]]) @ Wbig -> scatter-add into agg[dst[e],:]
// Never materializes the 1.6GB per-edge weight tensor; Wbig (139KB bf16) stays L2-resident.
__global__ void __launch_bounds__(128, 1)
k_edge_msg(const float* __restrict__ edge_x, const int* __restrict__ src,
           const int* __restrict__ dst, const float* __restrict__ hidden,
           const __bf16* __restrict__ wt, float* __restrict__ agg) {
    __shared__ float xs[128][16];           // 8KB
    __shared__ float hsrc[128][64];         // 32KB
    __shared__ __bf16 Ap[128 * LSTRIDE];    // 10KB  product tile, natural k order
    __shared__ __bf16 Bt[64 * LSTRIDE];     // 5KB   Wbig chunk, [o][k]
    __shared__ int srcl[128];
    __shared__ int dstl[128];

    const int tid = threadIdx.x;
    const int e0 = blockIdx.x * 128;

    {
        int e = e0 + tid;
        srcl[tid] = (e < EE) ? src[e] : -1;
        dstl[tid] = (e < EE) ? dst[e] : -1;
    }
    for (int i = tid; i < 128 * 16; i += 128) {
        int r = i >> 4, c = i & 15;
        int e = e0 + r;
        xs[r][c] = (e < EE) ? edge_x[(size_t)e * 16 + c] : 0.0f;
    }
    __syncthreads();
    for (int i = tid; i < 128 * 64; i += 128) {
        int r = i >> 6, c = i & 63;
        int s = srcl[r];
        hsrc[r][c] = (s >= 0) ? hidden[(size_t)s * 64 + c] : 0.0f;
    }
    __syncthreads();

    const int wv = tid >> 5, lane = tid & 31, hi = lane >> 4, mr = lane & 15;

    v8f acc[2][4];
#pragma unroll
    for (int f = 0; f < 2; ++f)
#pragma unroll
        for (int t = 0; t < 4; ++t)
#pragma unroll
            for (int r = 0; r < 8; ++r) acc[f][t][r] = 0.0f;

    for (int kc = 0; kc < NCHUNK; ++kc) {
        // ---- build A tile: U[e, gk] = x[e,gk/64]*h[e,gk%64] (gk<1024) else h[e,gk-1024]
        for (int i = tid; i < 128 * 8; i += 128) {
            int r = i >> 3, c4 = (i & 7) * 4;
            int gk = kc * 32 + c4;
            float a0, a1, a2, a3;
            if (gk < 1024) {
                float xv = xs[r][gk >> 6];
                const float* hp = &hsrc[r][gk & 63];
                a0 = xv * hp[0]; a1 = xv * hp[1]; a2 = xv * hp[2]; a3 = xv * hp[3];
            } else {
                const float* hp = &hsrc[r][gk - 1024];
                a0 = hp[0]; a1 = hp[1]; a2 = hp[2]; a3 = hp[3];
            }
            st4bf(Ap + r * LSTRIDE + c4, a0, a1, a2, a3);
        }
        // ---- stage Wbig chunk: wt is already [kc][o][k], so this is a straight 32B/thread copy
        {
            const __bf16* g = wt + (size_t)kc * 2048 + tid * 16;
            v8bf x0 = *(const v8bf*)(g);
            v8bf x1 = *(const v8bf*)(g + 8);
            __bf16* d = Bt + (tid >> 1) * LSTRIDE + (tid & 1) * 16;
            *(v8bf*)d = x0;
            *(v8bf*)(d + 8) = x1;
        }
        if (kc + 1 < NCHUNK)
            __builtin_prefetch(wt + (size_t)(kc + 1) * 2048 + tid * 16, 0, 1);
        __syncthreads();

        const __bf16* arow = Ap + (size_t)(wv * 32) * LSTRIDE + hi * 16;
        v16bf a0 = ld_frag(arow + (size_t)mr * LSTRIDE);
        v16bf a1 = ld_frag(arow + (size_t)(16 + mr) * LSTRIDE);
        v16bf b0 = ld_frag(Bt + (size_t)( 0 + mr) * LSTRIDE + hi * 16);
        v16bf b1 = ld_frag(Bt + (size_t)(16 + mr) * LSTRIDE + hi * 16);
        v16bf b2 = ld_frag(Bt + (size_t)(32 + mr) * LSTRIDE + hi * 16);
        v16bf b3 = ld_frag(Bt + (size_t)(48 + mr) * LSTRIDE + hi * 16);

        acc[0][0] = WMMA_BF16(a0, b0, acc[0][0]);
        acc[0][1] = WMMA_BF16(a0, b1, acc[0][1]);
        acc[0][2] = WMMA_BF16(a0, b2, acc[0][2]);
        acc[0][3] = WMMA_BF16(a0, b3, acc[0][3]);
        acc[1][0] = WMMA_BF16(a1, b0, acc[1][0]);
        acc[1][1] = WMMA_BF16(a1, b1, acc[1][1]);
        acc[1][2] = WMMA_BF16(a1, b2, acc[1][2]);
        acc[1][3] = WMMA_BF16(a1, b3, acc[1][3]);
        __syncthreads();
    }

    // scatter-add messages to destination nodes
#pragma unroll
    for (int f = 0; f < 2; ++f) {
#pragma unroll
        for (int t = 0; t < 4; ++t) {
#pragma unroll
            for (int r = 0; r < 8; ++r) {
                int e_row = wv * 32 + f * 16 + r + 8 * hi;
                int d = dstl[e_row];
                if (d >= 0) atomicAdd(&agg[(size_t)d * 64 + t * 16 + mr], acc[f][t][r]);
            }
        }
    }
}

// ---------------- pointwise kernels ----------------
__global__ void k_relu_bias(float* __restrict__ agg, const float* __restrict__ b) {
    long i = (long)blockIdx.x * blockDim.x + threadIdx.x;
    if (i >= (long)NN * 64) return;
    float v = agg[i] + b[i & 63];
    agg[i] = v > 0.0f ? v : 0.0f;
}

__global__ void k_gru(const float* __restrict__ gi, const float* __restrict__ gh,
                      float* __restrict__ hidden) {
    long i = (long)blockIdx.x * blockDim.x + threadIdx.x;
    if (i >= (long)NN * 64) return;
    long n = i >> 6; int j = (int)(i & 63);
    const float* gin = gi + n * 192;
    const float* ghn = gh + n * 192;
    float r  = sigm(gin[j]        + ghn[j]);
    float z  = sigm(gin[64 + j]   + ghn[64 + j]);
    float nn = tanhf(gin[128 + j] + r * ghn[128 + j]);
    float hp = hidden[i];
    hidden[i] = (1.0f - z) * nn + z * hp;
}

__global__ void k_concat(const float* __restrict__ h0, const float* __restrict__ h,
                         float* __restrict__ feat) {
    long i = (long)blockIdx.x * blockDim.x + threadIdx.x;
    if (i >= (long)NN * 64) return;
    long n = i >> 6; int j = (int)(i & 63);
    feat[n * 128 + j]      = h0[i];
    feat[n * 128 + 64 + j] = h[i];
}

__global__ void k_lstm(const float* __restrict__ gates, float* __restrict__ lc,
                       float* __restrict__ lh, float* __restrict__ q_star) {
    long i = (long)blockIdx.x * blockDim.x + threadIdx.x;
    if (i >= (long)GG * 128) return;
    long g = i >> 7; int d = (int)(i & 127);
    const float* gr = gates + g * 512;
    float ig = sigm(gr[d]);
    float fg = sigm(gr[128 + d]);
    float gg = tanhf(gr[256 + d]);
    float og = sigm(gr[384 + d]);
    float c = fg * lc[i] + ig * gg;
    lc[i] = c;
    float hq = og * tanhf(c);
    lh[i] = hq;
    q_star[g * 256 + d]       = hq;   // q half of q_star
    q_star[g * 256 + 128 + d] = 0.0f; // reset readout half for the scatter
}

__global__ void k_attn_dot(const float* __restrict__ feat, const float* __restrict__ q_star,
                           const int* __restrict__ n2g, float* __restrict__ evec,
                           unsigned* __restrict__ emax) {
    int n = blockIdx.x * blockDim.x + threadIdx.x;
    if (n >= NN) return;
    int g = n2g[n];
    const float* f = feat + (size_t)n * 128;
    const float* q = q_star + (size_t)g * 256;
    float s = 0.0f;
#pragma unroll 8
    for (int d = 0; d < 128; ++d) s += f[d] * q[d];
    evec[n] = s;
    atomicMax(&emax[g], fenc(s));
}

__global__ void k_attn_exp(const float* __restrict__ evec, const int* __restrict__ n2g,
                           const unsigned* __restrict__ emax, float* __restrict__ exv,
                           float* __restrict__ denom) {
    int n = blockIdx.x * blockDim.x + threadIdx.x;
    if (n >= NN) return;
    int g = n2g[n];
    float ev = __expf(evec[n] - fdec(emax[g]));
    exv[n] = ev;
    atomicAdd(&denom[g], ev);
}

__global__ void k_attn_scat(const float* __restrict__ feat, const float* __restrict__ exv,
                            const float* __restrict__ denom, const int* __restrict__ n2g,
                            float* __restrict__ q_star) {
    long i = (long)blockIdx.x * blockDim.x + threadIdx.x;
    if (i >= (long)NN * 128) return;
    long n = i >> 7; int d = (int)(i & 127);
    int g = n2g[n];
    float a = exv[n] / denom[g];
    atomicAdd(&q_star[(size_t)g * 256 + 128 + d], a * feat[i]);
}

// ---------------- host orchestration ----------------
static inline void launch_gemm(const float* A, const float* B, const float* bias, float* C,
                               int M, int N, int K, int transB, int act,
                               const float* alpha_p, int accum, hipStream_t s) {
    dim3 grid((N + 63) / 64, (M + 127) / 128);
    k_gemm<<<grid, dim3(128), 0, s>>>(A, B, bias, C, M, N, K, transB, act, alpha_p, accum);
}

extern "C" void kernel_launch(void* const* d_in, const int* in_sizes, int n_in,
                              void* d_out, int out_size, void* d_ws, size_t ws_size,
                              hipStream_t stream) {
    (void)in_sizes; (void)n_in; (void)out_size; (void)ws_size;
    const float* node_x   = (const float*)d_in[0];
    const float* edge_x   = (const float*)d_in[1];
    const int*   src      = (const int*)  d_in[2];
    const int*   dst      = (const int*)  d_in[3];
    const int*   n2g      = (const int*)  d_in[4];
    const float* proj_W   = (const float*)d_in[5];
    const float* proj_b   = (const float*)d_in[6];
    const float* edge_W   = (const float*)d_in[7];
    const float* edge_b   = (const float*)d_in[8];
    const float* conv_b   = (const float*)d_in[9];
    const float* gru_Wih  = (const float*)d_in[10];
    const float* gru_Whh  = (const float*)d_in[11];
    const float* gru_bih  = (const float*)d_in[12];
    const float* gru_bhh  = (const float*)d_in[13];
    const float* lstm_Wih = (const float*)d_in[14];
    const float* lstm_Whh = (const float*)d_in[15];
    const float* lstm_bih = (const float*)d_in[16];
    const float* lstm_bhh = (const float*)d_in[17];
    const float* sp_W     = (const float*)d_in[18];
    const float* sp_b     = (const float*)d_in[19];
    const float* prelu_a  = (const float*)d_in[20];
    float* out = (float*)d_out;

    // workspace carving (float units)
    float* w = (float*)d_ws;
    size_t off = 0;
    float* h0     = w + off; off += (size_t)NN * 64;
    float* hidden = w + off; off += (size_t)NN * 64;
    float* agg    = w + off; off += (size_t)NN * 64;
    float* gi     = w + off; off += (size_t)NN * 192;
    float* gh     = w + off; off += (size_t)NN * 192;
    float* feat   = w + off; off += (size_t)NN * 128;
    float* q_star = w + off; off += (size_t)GG * 256;   // q_star, lh, lc kept contiguous
    float* lh     = w + off; off += (size_t)GG * 128;
    float* lc     = w + off; off += (size_t)GG * 128;
    float* gates  = w + off; off += (size_t)GG * 512;
    float* evec   = w + off; off += (size_t)NN;
    float* exv    = w + off; off += (size_t)NN;
    float* denom  = w + off; off += (size_t)GG;         // denom + emax contiguous for one zero pass
    unsigned* emax = (unsigned*)(w + off); off += (size_t)GG;
    __bf16* wbig  = (__bf16*)(w + off);                 // KFUSED*64 bf16 (~139KB), chunk-transposed

    const int TB = 256;
    long nh = (long)NN * 64;

    // fused edge-weight matrix (34 chunks of [o=64][k=32], bf16) -- stays in L2
    k_build_wbig<<<(KFUSED * 64 + TB - 1) / TB, TB, 0, stream>>>(edge_W, edge_b, wbig);

    // h0 = relu(node_x @ proj_W + proj_b); hidden = h0
    launch_gemm(node_x, proj_W, proj_b, h0, NN, 64, 64, /*transB=*/0, /*relu*/1, nullptr, 0, stream);
    k_copy<<<(nh + TB - 1) / TB, TB, 0, stream>>>(hidden, h0, nh);

    // 3 message-passing steps
    for (int step = 0; step < 3; ++step) {
        k_zero<<<(nh + TB - 1) / TB, TB, 0, stream>>>(agg, nh);
        k_edge_msg<<<(EE + 127) / 128, 128, 0, stream>>>(edge_x, src, dst, hidden, wbig, agg);
        k_relu_bias<<<(nh + TB - 1) / TB, TB, 0, stream>>>(agg, conv_b);
        launch_gemm(agg,    gru_Wih, gru_bih, gi, NN, 192, 64, /*transB=*/1, 0, nullptr, 0, stream);
        launch_gemm(hidden, gru_Whh, gru_bhh, gh, NN, 192, 64, /*transB=*/1, 0, nullptr, 0, stream);
        k_gru<<<(nh + TB - 1) / TB, TB, 0, stream>>>(gi, gh, hidden);
    }

    // feat = [h0, h]
    k_concat<<<(nh + TB - 1) / TB, TB, 0, stream>>>(h0, hidden, feat);

    // Set2Set: zero q_star|lh|lc in one contiguous sweep (GG*(256+128+128) = GG*512 floats)
    k_zero<<<((long)GG * 512 + TB - 1) / TB, TB, 0, stream>>>(q_star, (long)GG * 512);
    for (int it = 0; it < 3; ++it) {
        launch_gemm(q_star, lstm_Wih, lstm_bih, gates, GG, 512, 256, /*transB=*/1, 0, nullptr, /*acc=*/0, stream);
        launch_gemm(lh,     lstm_Whh, lstm_bhh, gates, GG, 512, 128, /*transB=*/1, 0, nullptr, /*acc=*/1, stream);
        k_lstm<<<((long)GG * 128 + TB - 1) / TB, TB, 0, stream>>>(gates, lc, lh, q_star);
        k_zero<<<(2 * GG + TB - 1) / TB, TB, 0, stream>>>(denom, 2 * GG);  // denom + emax(enc 0)
        k_attn_dot<<<(NN + TB - 1) / TB, TB, 0, stream>>>(feat, q_star, n2g, evec, emax);
        k_attn_exp<<<(NN + TB - 1) / TB, TB, 0, stream>>>(evec, n2g, emax, exv, denom);
        k_attn_scat<<<((long)NN * 128 + TB - 1) / TB, TB, 0, stream>>>(feat, exv, denom, n2g, q_star);
    }

    // sparsify: y = prelu(q_star @ sp_W^T + sp_b)
    launch_gemm(q_star, sp_W, sp_b, out, GG, RR, 256, /*transB=*/1, /*prelu*/2, prelu_a, 0, stream);
}
